// UniteGCNLayer_32160715112879
// MI455X (gfx1250) — matrix operand order repaired and checked by reading, not compiled, not run on hardware
//
#include <hip/hip_runtime.h>

// ---------------------------------------------------------------------------
// UniteGCNLayer for MI455X (gfx1250, wave32, WMMA, async-to-LDS staging).
//   cast weights/x -> bf16
//   deg (atomic)
//   WMMA GEMMs (K templated, A tile staged to LDS via global_load_async_to_lds_b128):
//       q,k,v (bf16 out), skip/xWr/xWroot (f32 out)
//   edge logits (wave/edge, bf16 gather) + atomicMax segment-max
//   edge exp + atomicAdd segment-sum
//   edge aggregate (wave/edge): alpha*v, x, w*x  -> f32 atomic scatter
//   postagg: x1 token + neighbor means (bf16)
//   WMMA GEMMs: x2,x3 into token stack; qkv over 3N rows
//   wave-per-node attention (H=2, hd=64, seq=3)
//   fold Wproj+Wfc into Mcat[128,384]; final WMMA GEMM (K=384) -> d_out
// ---------------------------------------------------------------------------

#define DFEAT 128

typedef __bf16 bf16_t;
typedef bf16_t v16bf __attribute__((ext_vector_type(16)));
typedef float  v8f   __attribute__((ext_vector_type(8)));
typedef int    v4i_  __attribute__((ext_vector_type(4)));

union Frag32B { v16bf v; uint4 q[2]; };

__device__ __forceinline__ float bf2f(unsigned short u) {
  return __uint_as_float(((unsigned)u) << 16);
}
__device__ __forceinline__ unsigned short f2bf(float f) {
  unsigned u = __float_as_uint(f);
  u += 0x7FFFu + ((u >> 16) & 1u);           // round-to-nearest-even
  return (unsigned short)(u >> 16);
}
// monotonic float<->uint encoding for atomicMax-based segment max
__device__ __forceinline__ unsigned fenc(float f) {
  int i = __float_as_int(f);
  return (i >= 0) ? ((unsigned)i | 0x80000000u) : (unsigned)(~i);
}
__device__ __forceinline__ float fdec(unsigned u) {
  int i = (u & 0x80000000u) ? (int)(u & 0x7FFFFFFFu) : ~(int)u;
  return __int_as_float(i);
}

// ---- CDNA5 async global->LDS copy (ASYNCcnt-tracked); guarded w/ fallback ----
#if defined(__has_builtin)
#if __has_builtin(__builtin_amdgcn_global_load_async_to_lds_b128)
#define HAVE_ASYNC_LDS 1
#endif
#endif

__device__ __forceinline__ void async_copy_b128(const void* g, void* l) {
#ifdef HAVE_ASYNC_LDS
  // builtin takes v4i pointers: (global src, lds dst, imm offset, imm cpol).
  // generic LDS pointer low 32 bits == raw LDS offset (flat aperture rule).
  __builtin_amdgcn_global_load_async_to_lds_b128(
      (__attribute__((address_space(1))) v4i_*)(uintptr_t)g,
      (__attribute__((address_space(3))) v4i_*)(unsigned)(uintptr_t)l,
      0, 0);
#else
  *(uint4*)l = *(const uint4*)g;
#endif
}

__device__ __forceinline__ void async_copy_wait() {
#ifdef HAVE_ASYNC_LDS
#if __has_builtin(__builtin_amdgcn_s_wait_asynccnt)
  __builtin_amdgcn_s_wait_asynccnt(0);
#else
  asm volatile("s_wait_asynccnt 0" ::: "memory");
#endif
#endif
}

// ---------------- elementwise utility kernels ----------------

__global__ void fill_zero_kernel(float* p, long long n) {
  long long i = (long long)blockIdx.x * 256 + threadIdx.x;
  if (i < n) p[i] = 0.0f;
}

__global__ void cast_bf16_kernel(const float* __restrict__ in,
                                 unsigned short* __restrict__ out, long long n) {
  long long i = (long long)blockIdx.x * 256 + threadIdx.x;
  if (i < n) out[i] = f2bf(in[i]);
}

__global__ void degree_kernel(const int* __restrict__ dst, float* deg, int E) {
  int e = blockIdx.x * 256 + threadIdx.x;
  if (e < E) atomicAdd(&deg[dst[e]], 1.0f);
}

// ---------------- WMMA bf16 GEMM:  C[M,Nout] = A[M,K] * W[Nout,K]^T ----------
// A,W bf16 row-major. Optional f32 bias[Nout], f32 addend[M,Nout].
// Optional f32 out, optional bf16 out with (pitch,col0) to write token stacks.
// A tile (16 x K) staged block-wide into LDS with async B128 copies; A frags
// then come from LDS (ds_load_b128), B frags stream from global (L2-resident)
// in groups of 4 so loads clause together and WMMAs issue back-to-back.
// Fragment layouts used here (A/B contiguous-K order is this code's choice, not the ISA table):
//   A(16x32 bf16): lane = m + 16*(k/16), halves 0-15 = contiguous K
//   B(32x16 bf16): lane = n + 16*(k/16), halves 0-15 = contiguous K
//   C/D(16x16 f32): vgpr j, lanes 0-15 -> (m0+j, n), lanes 16-31 -> (m0+j+8, n)

__device__ __forceinline__ v16bf load_a_frag_lds(const unsigned short* At,
                                                 int lda, int k0, int lane) {
  const int m  = lane & 15;
  const int kb = k0 + ((lane >> 4) << 4);
  const unsigned short* p = At + (size_t)m * lda + kb;
  Frag32B f;
  f.q[0] = *(const uint4*)(p);        // K = kb .. kb+7
  f.q[1] = *(const uint4*)(p + 8);    // K = kb+8 .. kb+15
  return f.v;
}

__device__ __forceinline__ v16bf load_b_frag(const unsigned short* __restrict__ W,
                                             int ldw, int n0, int k0, int lane) {
  const int n  = n0 + (lane & 15);
  const int kb = k0 + ((lane >> 4) << 4);
  const unsigned short* p = W + (size_t)n * ldw + kb;
  Frag32B f;
  f.q[0] = *(const uint4*)(p);        // K = kb .. kb+7
  f.q[1] = *(const uint4*)(p + 8);    // K = kb+8 .. kb+15
  return f.v;
}

template <int K>
__global__ void wmma_gemm_kernel(const unsigned short* __restrict__ A,
                                 const unsigned short* __restrict__ W,
                                 const float* __restrict__ bias,
                                 const float* __restrict__ addend,
                                 float* __restrict__ outF,
                                 unsigned short* __restrict__ outB,
                                 int outB_pitch, int outB_col0, int Nout) {
  constexpr int KF = K / 32;                       // k-steps
  __shared__ __align__(16) unsigned short Atile[16 * K];

  const int lane = threadIdx.x & 31;
  const int wave = threadIdx.x >> 5;
  const int m0 = blockIdx.x << 4;                  // 16 rows per block

  // ---- stage A tile (16 x K bf16) into LDS: one async B128 per chunk ----
  constexpr int CHUNKS = (16 * K * 2) / (256 * 16);  // 1 (K=128) or 3 (K=384)
#pragma unroll
  for (int c = 0; c < CHUNKS; ++c) {
    const int elem = ((c << 8) + threadIdx.x) << 3;  // 8 ushorts = 16B
    async_copy_b128(A + (size_t)m0 * K + elem, Atile + elem);
  }
  async_copy_wait();
  __syncthreads();

  // ---- each wave preloads its A fragments once (reused across all N-tiles) --
  v16bf AF[KF];
#pragma unroll
  for (int kk = 0; kk < KF; ++kk)
    AF[kk] = load_a_frag_lds(Atile, K, kk * 32, lane);

  const int ntiles = Nout >> 4;
  for (int nt = wave; nt < ntiles; nt += 8) {
    const int n0 = nt << 4;
    v8f acc = {};
    constexpr int GRP = (KF < 4) ? KF : 4;           // B-frag prefetch depth
#pragma unroll
    for (int kg = 0; kg < KF; kg += GRP) {
      v16bf BF[GRP];
#pragma unroll
      for (int u = 0; u < GRP; ++u)                  // loads clause together
        BF[u] = load_b_frag(W, K, n0, (kg + u) * 32, lane);
#pragma unroll
      for (int u = 0; u < GRP; ++u)                  // WMMAs back-to-back
        acc = __builtin_amdgcn_wmma_f32_16x16x32_bf16(
            false, AF[kg + u], false, BF[u], (short)0, acc, false, false);
    }
    const int n = n0 + (lane & 15);
    const float bv = bias ? bias[n] : 0.0f;
    const int mbase = m0 + ((lane >> 4) << 3);
#pragma unroll
    for (int j = 0; j < 8; ++j) {
      const int m = mbase + j;
      float v = acc[j] + bv;
      if (addend) v += addend[(size_t)m * Nout + n];
      if (outF) outF[(size_t)m * Nout + n] = v;
      if (outB) outB[(size_t)m * outB_pitch + outB_col0 + n] = f2bf(v);
    }
  }
}

// ---------------- edge kernels (wave per edge) ----------------

__global__ void edge_logits_kernel(const int* __restrict__ src,
                                   const int* __restrict__ dst,
                                   const unsigned short* __restrict__ qb,
                                   const unsigned short* __restrict__ kb,
                                   float* __restrict__ logit,
                                   unsigned int* __restrict__ mEnc, int E) {
  const int tid = blockIdx.x * 256 + threadIdx.x;
  const int gw = tid >> 5;
  const int lane = threadIdx.x & 31;
  if (gw >= E) return;
  const int s = src[gw], d = dst[gw];
  const uint2 qv = *(const uint2*)(qb + (size_t)d * DFEAT + lane * 4);
  const uint2 kv = *(const uint2*)(kb + (size_t)s * DFEAT + lane * 4);
  float p = bf2f((unsigned short)(qv.x & 0xFFFF)) * bf2f((unsigned short)(kv.x & 0xFFFF))
          + bf2f((unsigned short)(qv.x >> 16))    * bf2f((unsigned short)(kv.x >> 16))
          + bf2f((unsigned short)(qv.y & 0xFFFF)) * bf2f((unsigned short)(kv.y & 0xFFFF))
          + bf2f((unsigned short)(qv.y >> 16))    * bf2f((unsigned short)(kv.y >> 16));
#pragma unroll
  for (int off = 16; off > 0; off >>= 1) p += __shfl_xor(p, off, 32);
  if (lane == 0) {
    const float lg = p * 0.08838834764831845f;  // 1/sqrt(128)
    logit[gw] = lg;
    atomicMax(&mEnc[d], fenc(lg));
  }
}

__global__ void edge_exp_kernel(const int* __restrict__ dst,
                                const float* __restrict__ logit,
                                const unsigned int* __restrict__ mEnc,
                                float* __restrict__ el,
                                float* __restrict__ sExp, int E) {
  int e = blockIdx.x * 256 + threadIdx.x;
  if (e >= E) return;
  const int d = dst[e];
  const float v = expf(logit[e] - fdec(mEnc[d]));
  el[e] = v;
  atomicAdd(&sExp[d], v);
}

__global__ void edge_agg_kernel(const int* __restrict__ src,
                                const int* __restrict__ dst,
                                const float* __restrict__ ew,
                                const float* __restrict__ el,
                                const float* __restrict__ sExp,
                                const unsigned short* __restrict__ vb,
                                const unsigned short* __restrict__ xb,
                                float* __restrict__ aggv,
                                float* __restrict__ aggx,
                                float* __restrict__ aggxw, int E) {
  const int tid = blockIdx.x * 256 + threadIdx.x;
  const int gw = tid >> 5;
  const int lane = threadIdx.x & 31;
  if (gw >= E) return;
  const int s = src[gw], d = dst[gw];
  const float alpha = el[gw] / sExp[d];
  const float w = ew[gw];
  const int c0 = lane * 4;
  const uint2 vv = *(const uint2*)(vb + (size_t)s * DFEAT + c0);
  const uint2 xv = *(const uint2*)(xb + (size_t)s * DFEAT + c0);
  float vf[4] = { bf2f((unsigned short)(vv.x & 0xFFFF)), bf2f((unsigned short)(vv.x >> 16)),
                  bf2f((unsigned short)(vv.y & 0xFFFF)), bf2f((unsigned short)(vv.y >> 16)) };
  float xf[4] = { bf2f((unsigned short)(xv.x & 0xFFFF)), bf2f((unsigned short)(xv.x >> 16)),
                  bf2f((unsigned short)(xv.y & 0xFFFF)), bf2f((unsigned short)(xv.y >> 16)) };
  float* av  = aggv  + (size_t)d * DFEAT + c0;
  float* ax  = aggx  + (size_t)d * DFEAT + c0;
  float* axw = aggxw + (size_t)d * DFEAT + c0;
#pragma unroll
  for (int j = 0; j < 4; ++j) {
    atomicAdd(av + j, alpha * vf[j]);
    atomicAdd(ax + j, xf[j]);
    atomicAdd(axw + j, w * xf[j]);
  }
}

// postagg: x1 token (aggv+skip), neighbor means in bf16
__global__ void postagg_kernel(const float* __restrict__ aggv,
                               const float* __restrict__ skipf,
                               const float* __restrict__ aggx,
                               const float* __restrict__ aggxw,
                               const float* __restrict__ deg,
                               unsigned short* __restrict__ xsb,   // [N, 3*128]
                               unsigned short* __restrict__ meanb,
                               unsigned short* __restrict__ meanwb,
                               long long total) {
  long long i = (long long)blockIdx.x * 256 + threadIdx.x;
  if (i >= total) return;
  const long long n = i >> 7;
  const int d = (int)(i & 127);
  const float dg = fmaxf(deg[n], 1.0f);
  xsb[n * 384 + d] = f2bf(aggv[i] + skipf[i]);
  meanb[i]  = f2bf(aggx[i] / dg);
  meanwb[i] = f2bf(aggxw[i] / dg);
}

// ---------------- per-node attention over the 3-token stack ----------------
// qkvb: bf16 [N*3, 384] rows = (n*3+t), cols = s*128 + h*64 + d (s in {q,k,v})
// attb: bf16 [N, 384],   cols = t*128 + h*64 + d
__global__ void node_attention_kernel(const unsigned short* __restrict__ qkvb,
                                      unsigned short* __restrict__ attb, int Nn) {
  const int tid = blockIdx.x * 256 + threadIdx.x;
  const int w = tid >> 5;
  const int lane = threadIdx.x & 31;
  if (w >= Nn) return;
  const size_t base = (size_t)w * 3 * 384;
  const float scale = 0.125f;  // 1/sqrt(64)
#pragma unroll
  for (int h = 0; h < 2; ++h) {
    float q0[3], q1[3], k0[3], k1[3], v0[3], v1[3];
#pragma unroll
    for (int t = 0; t < 3; ++t) {
      const size_t rb = base + (size_t)t * 384 + h * 64;
      q0[t] = bf2f(qkvb[rb + lane]);        q1[t] = bf2f(qkvb[rb + lane + 32]);
      k0[t] = bf2f(qkvb[rb + 128 + lane]);  k1[t] = bf2f(qkvb[rb + 128 + lane + 32]);
      v0[t] = bf2f(qkvb[rb + 256 + lane]);  v1[t] = bf2f(qkvb[rb + 256 + lane + 32]);
    }
    float s[3][3];
#pragma unroll
    for (int qi = 0; qi < 3; ++qi)
#pragma unroll
      for (int ki = 0; ki < 3; ++ki) {
        float p = q0[qi] * k0[ki] + q1[qi] * k1[ki];
#pragma unroll
        for (int off = 16; off > 0; off >>= 1) p += __shfl_xor(p, off, 32);
        s[qi][ki] = p * scale;
      }
#pragma unroll
    for (int qi = 0; qi < 3; ++qi) {
      const float mx = fmaxf(s[qi][0], fmaxf(s[qi][1], s[qi][2]));
      const float e0 = expf(s[qi][0] - mx), e1 = expf(s[qi][1] - mx), e2 = expf(s[qi][2] - mx);
      const float inv = 1.0f / (e0 + e1 + e2);
      const float a0 = e0 * inv, a1 = e1 * inv, a2 = e2 * inv;
      const size_t ob = (size_t)w * 384 + qi * 128 + h * 64;
      attb[ob + lane]      = f2bf(a0 * v0[0] + a1 * v0[1] + a2 * v0[2]);
      attb[ob + lane + 32] = f2bf(a0 * v1[0] + a1 * v1[1] + a2 * v1[2]);
    }
  }
}

// ---------------- fold Wproj+Wfc:  Mcat[o, t*128+i] = sum_j Wfc[o,t*128+j]*Wproj[j,i]
__global__ void mcat_kernel(const float* __restrict__ Wfc,
                            const float* __restrict__ Wproj,
                            unsigned short* __restrict__ mcatb) {
  const int idx = blockIdx.x * 256 + threadIdx.x;
  if (idx >= 128 * 384) return;
  const int o = idx / 384;
  const int c = idx % 384;
  const int t = c >> 7, i = c & 127;
  float acc = 0.0f;
  const float* wf = Wfc + (size_t)o * 384 + (t << 7);
#pragma unroll 4
  for (int j = 0; j < 128; ++j) acc += wf[j] * Wproj[(size_t)j * 128 + i];
  mcatb[(size_t)o * 384 + c] = f2bf(acc);
}

__global__ void biasc_kernel(const float* __restrict__ Wfc,
                             const float* __restrict__ bproj,
                             const float* __restrict__ bfc,
                             float* __restrict__ biasC) {
  const int o = threadIdx.x;
  if (o >= 128) return;
  float acc = bfc[o];
  for (int c = 0; c < 384; ++c) acc += Wfc[(size_t)o * 384 + c] * bproj[c & 127];
  biasC[o] = acc;
}

// ---------------------------------------------------------------------------

extern "C" void kernel_launch(void* const* d_in, const int* in_sizes, int n_in,
                              void* d_out, int out_size, void* d_ws, size_t ws_size,
                              hipStream_t stream) {
  const int N = in_sizes[0] / DFEAT;
  const int E = in_sizes[2];

  const float* x     = (const float*)d_in[0];
  const int*   ei    = (const int*)d_in[1];
  const float* ew    = (const float*)d_in[2];
  const float* Wq    = (const float*)d_in[3];
  const float* bq    = (const float*)d_in[4];
  const float* Wk    = (const float*)d_in[5];
  const float* bk    = (const float*)d_in[6];
  const float* Wv    = (const float*)d_in[7];
  const float* bv    = (const float*)d_in[8];
  const float* Wskip = (const float*)d_in[9];
  const float* bskip = (const float*)d_in[10];
  const float* Wl    = (const float*)d_in[11];
  const float* bl    = (const float*)d_in[12];
  const float* Wr    = (const float*)d_in[13];
  const float* Wrel  = (const float*)d_in[14];
  const float* brel  = (const float*)d_in[15];
  const float* Wroot = (const float*)d_in[16];
  const float* Wqkv  = (const float*)d_in[17];
  const float* bqkv  = (const float*)d_in[18];
  const float* Wproj = (const float*)d_in[19];
  const float* bproj = (const float*)d_in[20];
  const float* Wfc   = (const float*)d_in[21];
  const float* bfc   = (const float*)d_in[22];

  const int* src = ei;
  const int* dst = ei + E;

  // ---- workspace carve-up (256B aligned) ----
  char* p = (char*)d_ws;
  auto alloc = [&](size_t bytes) -> void* {
    void* r = (void*)p;
    p += (bytes + 255) & ~(size_t)255;
    return r;
  };
  const size_t ND = (size_t)N * DFEAT;

  unsigned short* wbq    = (unsigned short*)alloc(128 * 128 * 2);
  unsigned short* wbk    = (unsigned short*)alloc(128 * 128 * 2);
  unsigned short* wbv    = (unsigned short*)alloc(128 * 128 * 2);
  unsigned short* wbskip = (unsigned short*)alloc(128 * 128 * 2);
  unsigned short* wbl    = (unsigned short*)alloc(128 * 128 * 2);
  unsigned short* wbr    = (unsigned short*)alloc(128 * 128 * 2);
  unsigned short* wbrel  = (unsigned short*)alloc(128 * 128 * 2);
  unsigned short* wbroot = (unsigned short*)alloc(128 * 128 * 2);
  unsigned short* wbqkv  = (unsigned short*)alloc(384 * 128 * 2);

  unsigned short* xb = (unsigned short*)alloc(ND * 2);
  unsigned short* qb = (unsigned short*)alloc(ND * 2);
  unsigned short* kb = (unsigned short*)alloc(ND * 2);
  unsigned short* vb = (unsigned short*)alloc(ND * 2);
  float* skipf  = (float*)alloc(ND * 4);
  float* xrf    = (float*)alloc(ND * 4);
  float* xrootf = (float*)alloc(ND * 4);

  char* z0 = p;                    // zero-initialized region start
  float*        deg   = (float*)alloc((size_t)N * 4);
  unsigned int* mEnc  = (unsigned int*)alloc((size_t)N * 4);
  float*        sExp  = (float*)alloc((size_t)N * 4);
  float*        aggv  = (float*)alloc(ND * 4);
  float*        aggx  = (float*)alloc(ND * 4);
  float*        aggxw = (float*)alloc(ND * 4);
  const long long zcount = (long long)((p - z0) / 4);

  float* logit = (float*)alloc((size_t)E * 4);
  float* el    = (float*)alloc((size_t)E * 4);
  unsigned short* meanb  = (unsigned short*)alloc(ND * 2);
  unsigned short* meanwb = (unsigned short*)alloc(ND * 2);
  unsigned short* xsb    = (unsigned short*)alloc((size_t)N * 384 * 2);
  unsigned short* qkvb   = (unsigned short*)alloc((size_t)N * 3 * 384 * 2);
  unsigned short* attb   = (unsigned short*)alloc((size_t)N * 384 * 2);
  unsigned short* mcatb  = (unsigned short*)alloc(128 * 384 * 2);
  float* biasC = (float*)alloc(128 * 4);
  (void)ws_size; (void)n_in; (void)out_size;

  auto nb = [](long long t) -> unsigned { return (unsigned)((t + 255) / 256); };

  // ---- phase 0: init + casts ----
  fill_zero_kernel<<<nb(zcount), 256, 0, stream>>>((float*)z0, zcount);
  cast_bf16_kernel<<<nb(16384), 256, 0, stream>>>(Wq, wbq, 16384);
  cast_bf16_kernel<<<nb(16384), 256, 0, stream>>>(Wk, wbk, 16384);
  cast_bf16_kernel<<<nb(16384), 256, 0, stream>>>(Wv, wbv, 16384);
  cast_bf16_kernel<<<nb(16384), 256, 0, stream>>>(Wskip, wbskip, 16384);
  cast_bf16_kernel<<<nb(16384), 256, 0, stream>>>(Wl, wbl, 16384);
  cast_bf16_kernel<<<nb(16384), 256, 0, stream>>>(Wr, wbr, 16384);
  cast_bf16_kernel<<<nb(16384), 256, 0, stream>>>(Wrel, wbrel, 16384);
  cast_bf16_kernel<<<nb(16384), 256, 0, stream>>>(Wroot, wbroot, 16384);
  cast_bf16_kernel<<<nb(49152), 256, 0, stream>>>(Wqkv, wbqkv, 49152);
  cast_bf16_kernel<<<nb((long long)ND), 256, 0, stream>>>(x, xb, (long long)ND);
  degree_kernel<<<nb(E), 256, 0, stream>>>(dst, deg, E);

  // ---- phase 1: six projections of x (WMMA, K=128) ----
  const unsigned gemmN = (unsigned)(N / 16);
  wmma_gemm_kernel<128><<<gemmN, 256, 0, stream>>>(xb, wbq, bq, nullptr, nullptr, qb, 128, 0, 128);
  wmma_gemm_kernel<128><<<gemmN, 256, 0, stream>>>(xb, wbk, bk, nullptr, nullptr, kb, 128, 0, 128);
  wmma_gemm_kernel<128><<<gemmN, 256, 0, stream>>>(xb, wbv, bv, nullptr, nullptr, vb, 128, 0, 128);
  wmma_gemm_kernel<128><<<gemmN, 256, 0, stream>>>(xb, wbskip, bskip, nullptr, skipf, nullptr, 0, 0, 128);
  wmma_gemm_kernel<128><<<gemmN, 256, 0, stream>>>(xb, wbr, nullptr, nullptr, xrf, nullptr, 0, 0, 128);
  wmma_gemm_kernel<128><<<gemmN, 256, 0, stream>>>(xb, wbroot, nullptr, nullptr, xrootf, nullptr, 0, 0, 128);

  // ---- phase 2: edge softmax-attention + aggregations ----
  edge_logits_kernel<<<nb((long long)E * 32), 256, 0, stream>>>(src, dst, qb, kb, logit, mEnc, E);
  edge_exp_kernel<<<nb(E), 256, 0, stream>>>(dst, logit, mEnc, el, sExp, E);
  edge_agg_kernel<<<nb((long long)E * 32), 256, 0, stream>>>(src, dst, ew, el, sExp, vb, xb,
                                                             aggv, aggx, aggxw, E);
  postagg_kernel<<<nb((long long)ND), 256, 0, stream>>>(aggv, skipf, aggx, aggxw, deg,
                                                        xsb, meanb, meanwb, (long long)ND);

  // ---- phase 3: x2, x3 tokens (WMMA, write bf16 into token stack) ----
  wmma_gemm_kernel<128><<<gemmN, 256, 0, stream>>>(meanb,  wbl,   bl,   xrf,    nullptr, xsb, 384, 128, 128);
  wmma_gemm_kernel<128><<<gemmN, 256, 0, stream>>>(meanwb, wbrel, brel, xrootf, nullptr, xsb, 384, 256, 128);

  // ---- phase 4: qkv over 3N token rows (WMMA), then per-node attention ----
  wmma_gemm_kernel<128><<<(unsigned)(3 * N / 16), 256, 0, stream>>>(xsb, wbqkv, bqkv, nullptr,
                                                                    nullptr, qkvb, 384, 0, 384);
  node_attention_kernel<<<nb((long long)N * 32), 256, 0, stream>>>(qkvb, attb, N);

  // ---- phase 5: folded Wproj*Wfc, final WMMA GEMM (K=384) -> d_out ----
  mcat_kernel<<<nb(128 * 384), 256, 0, stream>>>(Wfc, Wproj, mcatb);
  biasc_kernel<<<1, 128, 0, stream>>>(Wfc, bproj, bfc, biasC);
  wmma_gemm_kernel<384><<<gemmN, 256, 0, stream>>>(attb, mcatb, biasC, nullptr,
                                                   (float*)d_out, nullptr, 0, 0, 128);
}
